// Attention_35201551958191
// MI455X (gfx1250) — compile-verified, matrix operand order checked by reading, not run on hardware
//
#include <hip/hip_runtime.h>

typedef __bf16 bf16x16 __attribute__((ext_vector_type(16)));
typedef float  v8f     __attribute__((ext_vector_type(8)));
typedef unsigned short ushort_t;

union AFrag {
    bf16x16 v;
    uint4   q[2];
};

__device__ __forceinline__ ushort_t f2bf(float f) {
    unsigned int u = __float_as_uint(f);
    unsigned int r = (u + 0x7FFFu + ((u >> 16) & 1u)) >> 16;
    return (ushort_t)r;
}

__device__ __forceinline__ v8f wmma_bf16(bf16x16 a, bf16x16 b, v8f c) {
    return __builtin_amdgcn_wmma_f32_16x16x32_bf16(false, a, false, b, (short)0, c, false, false);
}

// CDNA5 async global->LDS copy (ASYNCcnt-tracked), bypasses VGPR staging.
// lds_off: byte offset in LDS (low 32 bits of a shared-memory generic pointer
// are exactly the LDS offset per the gfx1250 aperture layout).
__device__ __forceinline__ void async_copy_b128(void* lds_ptr, const void* gptr) {
    unsigned lds_off = (unsigned)(uintptr_t)lds_ptr;
    unsigned long long ga = (unsigned long long)(uintptr_t)gptr;
    asm volatile("global_load_async_to_lds_b128 %0, %1, off"
                 :: "v"(lds_off), "v"(ga) : "memory");
}
__device__ __forceinline__ void wait_async_le4() {
    asm volatile("s_wait_asynccnt 0x4" ::: "memory");
}
__device__ __forceinline__ void wait_async_0() {
    asm volatile("s_wait_asynccnt 0x0" ::: "memory");
}

// ---------------------------------------------------------------------------
// fp32 (R x C, row-major) -> bf16 TRANSPOSED (C x R, row-major)
// ---------------------------------------------------------------------------
__global__ void cvt_f32_bf16_t(const float* __restrict__ in, ushort_t* __restrict__ out,
                               int R, int C) {
    int idx = blockIdx.x * 256 + threadIdx.x;
    if (idx < R * C) {
        int r = idx / C, c = idx % C;
        out[(size_t)c * R + r] = f2bf(in[idx]);
    }
}

// ---------------------------------------------------------------------------
// LayerNorm: one row (1024) per block of 256 threads -> bf16
// ---------------------------------------------------------------------------
__global__ void layernorm_k(const float* __restrict__ x, const float* __restrict__ w,
                            const float* __restrict__ bia, ushort_t* __restrict__ xn) {
    int row = blockIdx.x;
    int tid = threadIdx.x;
    const float* xr = x + (size_t)row * 1024;
    float vals[4];
    float s = 0.f, s2 = 0.f;
#pragma unroll
    for (int i = 0; i < 4; ++i) {
        float t = xr[tid + i * 256];
        vals[i] = t; s += t; s2 += t * t;
    }
#pragma unroll
    for (int off = 1; off < 32; off <<= 1) {
        s  += __shfl_xor(s,  off, 32);
        s2 += __shfl_xor(s2, off, 32);
    }
    __shared__ float rs[8], rs2[8];
    int wid = tid >> 5, lane = tid & 31;
    if (lane == 0) { rs[wid] = s; rs2[wid] = s2; }
    __syncthreads();
    if (wid == 0) {
        float a  = (lane < 8) ? rs[lane]  : 0.f;
        float a2 = (lane < 8) ? rs2[lane] : 0.f;
#pragma unroll
        for (int off = 1; off < 8; off <<= 1) {
            a  += __shfl_xor(a,  off, 32);
            a2 += __shfl_xor(a2, off, 32);
        }
        if (lane == 0) { rs[0] = a; rs2[0] = a2; }
    }
    __syncthreads();
    float mu   = rs[0] * (1.0f / 1024.0f);
    float var  = rs2[0] * (1.0f / 1024.0f) - mu * mu;
    float rstd = rsqrtf(var + 1e-5f);
#pragma unroll
    for (int i = 0; i < 4; ++i) {
        int c = tid + i * 256;
        float t = (vals[i] - mu) * rstd * w[c] + bia[c];
        xn[(size_t)row * 1024 + c] = f2bf(t);
    }
}

// ---------------------------------------------------------------------------
// Tiled bf16 GEMM: C(f32, MxN) = A(bf16, MxK row-major) * Bt(bf16, NxK row-major)^T
// Block tile 128x128, BK=32, 8 waves (2x4), each wave 64x32 (4x2 WMMA tiles).
// Double-buffered LDS filled with async global->LDS B128 copies.
// ---------------------------------------------------------------------------
#define LDS_STRIDE 40   // bf16 elems per LDS row (32 used + pad): conflict-free

__global__ void gemm_bf16_f32(const ushort_t* __restrict__ A, const ushort_t* __restrict__ Bt,
                              float* __restrict__ C, int M, int N, int K) {
    __shared__ ushort_t sA[2][128 * LDS_STRIDE];
    __shared__ ushort_t sB[2][128 * LDS_STRIDE];   // sB[n][k]
    int bm = blockIdx.y, bn = blockIdx.x;
    int tid = threadIdx.x;
    int wid = tid >> 5, lane = tid & 31;
    int wm = wid >> 2, wn = wid & 3;
    int l15 = lane & 15, half = lane >> 4;
    int kb = half * 8;

    v8f acc[4][2];
#pragma unroll
    for (int mi = 0; mi < 4; ++mi)
#pragma unroll
        for (int ni = 0; ni < 2; ++ni) acc[mi][ni] = (v8f){0,0,0,0,0,0,0,0};

    // per-thread tile-fill coordinates (A and Bt share the pattern: 128 rows x 32 k)
    auto issue_tile = [&](int buf, int k0) {
#pragma unroll
        for (int i = 0; i < 2; ++i) {
            int e = (tid + i * 256) * 8;
            int r = e >> 5, c = e & 31;
            async_copy_b128(sA[buf] + r * LDS_STRIDE + c,
                            A  + (size_t)(bm * 128 + r) * K + k0 + c);
            async_copy_b128(sB[buf] + r * LDS_STRIDE + c,
                            Bt + (size_t)(bn * 128 + r) * K + k0 + c);
        }
    };

    const int nk = K >> 5;
    issue_tile(0, 0);
    for (int kt = 0; kt < nk; ++kt) {
        int cur = kt & 1;
        if (kt + 1 < nk) {
            issue_tile(cur ^ 1, (kt + 1) * 32);
            wait_async_le4();   // previous group (this buf) complete; next stays in flight
        } else {
            wait_async_0();
        }
        __syncthreads();

        AFrag bfr[2];
#pragma unroll
        for (int ni = 0; ni < 2; ++ni) {
            const ushort_t* bp = sB[cur] + (wn * 32 + ni * 16 + l15) * LDS_STRIDE + kb;
            bfr[ni].q[0] = *(const uint4*)(bp);
            bfr[ni].q[1] = *(const uint4*)(bp + 16);
        }
#pragma unroll
        for (int mi = 0; mi < 4; ++mi) {
            AFrag af;
            const ushort_t* ap = sA[cur] + (wm * 64 + mi * 16 + l15) * LDS_STRIDE + kb;
            af.q[0] = *(const uint4*)(ap);
            af.q[1] = *(const uint4*)(ap + 16);
#pragma unroll
            for (int ni = 0; ni < 2; ++ni)
                acc[mi][ni] = wmma_bf16(af.v, bfr[ni].v, acc[mi][ni]);
        }
        __syncthreads();
    }

#pragma unroll
    for (int mi = 0; mi < 4; ++mi)
#pragma unroll
        for (int ni = 0; ni < 2; ++ni) {
            float* cp = C + (size_t)(bm * 128 + wm * 64 + mi * 16 + half * 8) * N
                          + bn * 128 + wn * 32 + ni * 16 + l15;
#pragma unroll
            for (int v = 0; v < 8; ++v)
                cp[(size_t)v * N] = acc[mi][ni][v];
        }
}

// ---------------------------------------------------------------------------
// L2-normalize q per (token, head), fold in softmax SCALE=10; out [b][h][n][64]
// ---------------------------------------------------------------------------
__global__ void l2norm_q_k(const float* __restrict__ q32, ushort_t* __restrict__ qbf) {
    int gw   = (blockIdx.x * 256 + threadIdx.x) >> 5;  // token*16 + head
    int lane = threadIdx.x & 31;
    int row  = gw >> 4;
    int h    = gw & 15;
    const float* p = q32 + (size_t)row * 1024 + h * 64;
    float a = p[lane * 2], b = p[lane * 2 + 1];
    float ss = a * a + b * b;
#pragma unroll
    for (int off = 1; off < 32; off <<= 1) ss += __shfl_xor(ss, off, 32);
    float sc = 10.0f / fmaxf(sqrtf(ss), 1e-12f);   // SCALE folded in
    int bb = row >> 11, i = row & 2047;
    ushort_t* o = qbf + (((size_t)(bb * 16 + h) * 2048) + i) * 64;
    o[lane * 2]     = f2bf(a * sc);
    o[lane * 2 + 1] = f2bf(b * sc);
}

// ---------------------------------------------------------------------------
// Split kv -> L2-normalized k (bf16) and v (bf16). One wave per token.
// ---------------------------------------------------------------------------
__global__ void l2norm_kv_k(const float* __restrict__ kv32,
                            ushort_t* __restrict__ kbf, ushort_t* __restrict__ vbf) {
    int tok  = (blockIdx.x * 256 + threadIdx.x) >> 5;
    int lane = threadIdx.x & 31;
    const float* p = kv32 + (size_t)tok * 128;
    float a = p[lane * 2], b = p[lane * 2 + 1];
    float ss = a * a + b * b;
#pragma unroll
    for (int off = 1; off < 32; off <<= 1) ss += __shfl_xor(ss, off, 32);
    float sc = 1.0f / fmaxf(sqrtf(ss), 1e-12f);
    kbf[(size_t)tok * 64 + lane * 2]     = f2bf(a * sc);
    kbf[(size_t)tok * 64 + lane * 2 + 1] = f2bf(b * sc);
    float va = p[64 + lane * 2], vb = p[64 + lane * 2 + 1];
    vbf[(size_t)tok * 64 + lane * 2]     = f2bf(va);
    vbf[(size_t)tok * 64 + lane * 2 + 1] = f2bf(vb);
}

// ---------------------------------------------------------------------------
// Causal flash attention. Grid: (n/128, b*heads). Block 256 = 8 waves.
// ---------------------------------------------------------------------------
#define MASKV (-3.0e38f)

__global__ void flash_attn_k(const ushort_t* __restrict__ qbf,
                             const ushort_t* __restrict__ kbf,
                             const ushort_t* __restrict__ vbf,
                             const unsigned char* __restrict__ cmask,
                             ushort_t* __restrict__ aobf) {
    int qblk = blockIdx.x;
    int bh   = blockIdx.y;
    int b    = bh >> 4, h = bh & 15;
    int tid  = threadIdx.x, wid = tid >> 5, lane = tid & 31;
    int l15  = lane & 15, half = lane >> 4;
    int kb   = half * 8;
    int q0   = qblk * 128 + wid * 16;

    __shared__ ushort_t sK[32 * 72];      // K tile [32 keys][64 dims], pad 72
    __shared__ ushort_t sVT[64 * 40];     // V^T tile [64 dims][32 keys], pad 40
    __shared__ ushort_t sP[8][16 * 40];   // per-wave P tile [16 rows][32 keys]

    AFrag qf[2];
    {
        const ushort_t* qbase = qbf + ((size_t)bh * 2048 + q0 + l15) * 64;
#pragma unroll
        for (int kk = 0; kk < 2; ++kk) {
            qf[kk].q[0] = *(const uint4*)(qbase + kk * 32 + kb);
            qf[kk].q[1] = *(const uint4*)(qbase + kk * 32 + kb + 16);
        }
    }

    v8f acc[4];
#pragma unroll
    for (int ot = 0; ot < 4; ++ot) acc[ot] = (v8f){0,0,0,0,0,0,0,0};
    float mrow[8], lrow[8];
#pragma unroll
    for (int v = 0; v < 8; ++v) { mrow[v] = MASKV; lrow[v] = 0.f; }

    const ushort_t* Kbase = kbf + (size_t)b * 2048 * 64;
    const ushort_t* Vbase = vbf + (size_t)b * 2048 * 64;
    const unsigned char* mb = cmask + (size_t)b * 2048;

    int nsteps    = qblk * 4 + 4;
    int wave_jmax = q0 + 15;
    int le = tid * 8;
    int lr = le >> 6, lc = le & 63;       // this thread's tile-fill coords

    for (int jt = 0; jt < nsteps; ++jt) {
        int j0 = jt * 32;
        // K tile: async straight copy; V tile: load + transpose scatter
        async_copy_b128(sK + lr * 72 + lc, Kbase + (size_t)(j0 + lr) * 64 + lc);
        {
            uint4 dv = *(const uint4*)(Vbase + (size_t)(j0 + lr) * 64 + lc);
            ushort_t tmp[8];
            *(uint4*)tmp = dv;
#pragma unroll
            for (int j = 0; j < 8; ++j) sVT[(lc + j) * 40 + lr] = tmp[j];
        }
        if (jt + 1 < nsteps) {  // warm L2 one tile ahead
            __builtin_prefetch(Kbase + (size_t)(j0 + 32 + lr) * 64 + lc, 0, 1);
            __builtin_prefetch(Vbase + (size_t)(j0 + 32 + lr) * 64 + lc, 0, 1);
        }
        wait_async_0();
        __syncthreads();

        if (j0 <= wave_jmax) {
            v8f s[2];
#pragma unroll
            for (int nt = 0; nt < 2; ++nt) {
                const ushort_t* kp = sK + (nt * 16 + l15) * 72;
                AFrag b0, b1;
                b0.q[0] = *(const uint4*)(kp + kb);
                b0.q[1] = *(const uint4*)(kp + kb + 16);
                b1.q[0] = *(const uint4*)(kp + 32 + kb);
                b1.q[1] = *(const uint4*)(kp + 32 + kb + 16);
                v8f t = (v8f){0,0,0,0,0,0,0,0};
                t = wmma_bf16(qf[0].v, b0.v, t);
                t = wmma_bf16(qf[1].v, b1.v, t);
                s[nt] = t;
            }
            float cand[8];
#pragma unroll
            for (int v = 0; v < 8; ++v) cand[v] = MASKV;
#pragma unroll
            for (int nt = 0; nt < 2; ++nt) {
                int j = j0 + nt * 16 + l15;
                bool cm = mb[j] != 0;
#pragma unroll
                for (int v = 0; v < 8; ++v) {
                    int i = q0 + half * 8 + v;
                    float val = s[nt][v];           // scale pre-folded into q
                    if (cm || (j > i)) val = MASKV;
                    s[nt][v] = val;
                    cand[v] = fmaxf(cand[v], val);
                }
            }
#pragma unroll
            for (int off = 1; off < 16; off <<= 1)
#pragma unroll
                for (int v = 0; v < 8; ++v)
                    cand[v] = fmaxf(cand[v], __shfl_xor(cand[v], off, 32));

            float alpha[8], psum[8];
#pragma unroll
            for (int v = 0; v < 8; ++v) {
                float mn = fmaxf(mrow[v], cand[v]);
                alpha[v] = __expf(mrow[v] - mn);
                mrow[v]  = mn;
                psum[v]  = 0.f;
            }
#pragma unroll
            for (int nt = 0; nt < 2; ++nt) {
#pragma unroll
                for (int v = 0; v < 8; ++v) {
                    float p = __expf(s[nt][v] - mrow[v]);
                    psum[v] += p;
                    sP[wid][(half * 8 + v) * 40 + nt * 16 + l15] = f2bf(p);
                }
            }
#pragma unroll
            for (int off = 1; off < 16; off <<= 1)
#pragma unroll
                for (int v = 0; v < 8; ++v)
                    psum[v] += __shfl_xor(psum[v], off, 32);
#pragma unroll
            for (int v = 0; v < 8; ++v) lrow[v] = lrow[v] * alpha[v] + psum[v];
#pragma unroll
            for (int ot = 0; ot < 4; ++ot)
#pragma unroll
                for (int v = 0; v < 8; ++v) acc[ot][v] *= alpha[v];

            AFrag pf;
            const ushort_t* pp = &sP[wid][l15 * 40 + kb];
            pf.q[0] = *(const uint4*)(pp);
            pf.q[1] = *(const uint4*)(pp + 16);
#pragma unroll
            for (int ot = 0; ot < 4; ++ot) {
                AFrag vf;
                const ushort_t* vp = sVT + (ot * 16 + l15) * 40 + kb;
                vf.q[0] = *(const uint4*)(vp);
                vf.q[1] = *(const uint4*)(vp + 16);
                acc[ot] = wmma_bf16(pf.v, vf.v, acc[ot]);
            }
        }
        __syncthreads();
    }

#pragma unroll
    for (int v = 0; v < 8; ++v) lrow[v] = (lrow[v] > 0.f) ? (1.0f / lrow[v]) : 0.f;
#pragma unroll
    for (int ot = 0; ot < 4; ++ot) {
        int col = h * 64 + ot * 16 + l15;
#pragma unroll
        for (int v = 0; v < 8; ++v) {
            int i = q0 + half * 8 + v;
            aobf[((size_t)b * 2048 + i) * 1024 + col] = f2bf(acc[ot][v] * lrow[v]);
        }
    }
}

// ---------------------------------------------------------------------------
// Host launch
// ---------------------------------------------------------------------------
extern "C" void kernel_launch(void* const* d_in, const int* in_sizes, int n_in,
                              void* d_out, int out_size, void* d_ws, size_t ws_size,
                              hipStream_t stream) {
    const float*         x     = (const float*)d_in[0];
    const unsigned char* cmask = (const unsigned char*)d_in[1];
    const float*         ln_w  = (const float*)d_in[2];
    const float*         ln_b  = (const float*)d_in[3];
    const float*         Wq    = (const float*)d_in[4];
    const float*         Wkv   = (const float*)d_in[5];
    const float*         Wo    = (const float*)d_in[6];
    float*               out   = (float*)d_out;

    const int B = 4, N = 2048, D = 1024, H = 16, DH = 64;
    const int M = B * N;  // 8192

    char* ws = (char*)d_ws;
    ushort_t* xn_bf   = (ushort_t*)ws;  ws += (size_t)M * D * 2;     // 16 MB
    ushort_t* WqT_bf  = (ushort_t*)ws;  ws += (size_t)D * D * 2;     // 2 MB   [N][K]
    ushort_t* WkvT_bf = (ushort_t*)ws;  ws += (size_t)D * 128 * 2;   // 256 KB [128][K]
    ushort_t* WoT_bf  = (ushort_t*)ws;  ws += (size_t)D * D * 2;     // 2 MB   [N][K]
    float*    q32     = (float*)ws;     ws += (size_t)M * D * 4;     // 32 MB
    float*    kv32    = (float*)ws;     ws += (size_t)M * 128 * 4;   // 4 MB
    ushort_t* qbf     = (ushort_t*)ws;  ws += (size_t)M * D * 2;     // 16 MB
    ushort_t* kbf     = (ushort_t*)ws;  ws += (size_t)M * DH * 2;    // 1 MB
    ushort_t* vbf     = (ushort_t*)ws;  ws += (size_t)M * DH * 2;    // 1 MB
    ushort_t* aobf    = (ushort_t*)ws;  ws += (size_t)M * D * 2;     // 16 MB

    // 1. weights -> bf16, transposed to [N][K] so GEMM tile fills are straight copies
    cvt_f32_bf16_t<<<(D * D + 255) / 256, 256, 0, stream>>>(Wq,  WqT_bf,  D, D);
    cvt_f32_bf16_t<<<(D * 128 + 255) / 256, 256, 0, stream>>>(Wkv, WkvT_bf, D, 128);
    cvt_f32_bf16_t<<<(D * D + 255) / 256, 256, 0, stream>>>(Wo,  WoT_bf,  D, D);

    // 2. LayerNorm -> bf16 activations
    layernorm_k<<<M, 256, 0, stream>>>(x, ln_w, ln_b, xn_bf);

    // 3. projections
    gemm_bf16_f32<<<dim3(D / 128, M / 128), 256, 0, stream>>>(xn_bf, WqT_bf,  q32,  M, D,   D);
    gemm_bf16_f32<<<dim3(1,       M / 128), 256, 0, stream>>>(xn_bf, WkvT_bf, kv32, M, 128, D);

    // 4. L2 norms (q gets SCALE folded in)
    l2norm_q_k<<<(M * H) / 8, 256, 0, stream>>>(q32, qbf);
    l2norm_kv_k<<<M / 8, 256, 0, stream>>>(kv32, kbf, vbf);

    // 5. causal flash attention
    flash_attn_k<<<dim3(N / 128, B * H), 256, 0, stream>>>(qbf, kbf, vbf, cmask, aobf);

    // 6. output projection
    gemm_bf16_f32<<<dim3(D / 128, M / 128), 256, 0, stream>>>(aobf, WoT_bf, out, M, D, D);
}
